// DebateGraph_42932493090927
// MI455X (gfx1250) — matrix operand order repaired and checked by reading, not compiled, not run on hardware
//
#include <hip/hip_runtime.h>
#include <hip/hip_bf16.h>
#include <math.h>

// ---------------- problem constants (match reference) ----------------
#define NN_    12288   // B*T*N nodes
#define NPG_   384     // nodes per graph (T*N)
#define NHID_  256
#define NFEAT_ 768
#define DEG_   32
#define H_     4
#define B_     32
#define T_     6
#define ACT_   2048    // active nodes per turn (B * N)
#define SLOPE_ 0.2f
#define BSTRIDE_ 40    // halves per LDS row: 32 data + 4-dword TDM pad (bank-conflict-free)

typedef __attribute__((ext_vector_type(16))) _Float16     v16h;
typedef __attribute__((ext_vector_type(8)))  _Float16     v8h;
typedef __attribute__((ext_vector_type(8)))  float        v8f;
typedef __attribute__((ext_vector_type(4)))  unsigned int u32x4;
typedef __attribute__((ext_vector_type(8)))  int          i32x8;
typedef __attribute__((ext_vector_type(4)))  int          i32x4;

__device__ __forceinline__ _Float16 f2h(float x) { return (_Float16)x; }

// =====================================================================
// TDM: tensor_load_to_lds of a 256-row x 32-element f16 tile of B^T.
//   dim0 (contiguous) = K direction (32 elems), dim1 = 256 N-rows,
//   tensor_dim0_stride = K. LDS pad: +4 dwords every 16 dwords ->
//   40-half row stride (16B-aligned rows, conflict-free half-wave b128).
// D# built per CDNA5 ISA.
// =====================================================================
__device__ __forceinline__ void tdm_load_slab(const _Float16* gptr,
                                              unsigned lds_off, int strideK)
{
    unsigned long long ga = (unsigned long long)(uintptr_t)gptr;
    u32x4 g0;
    g0[0] = 1u;                                        // count=1 (valid, user)
    g0[1] = lds_off;                                   // lds_addr [63:32]
    g0[2] = (unsigned)(ga & 0xffffffffu);              // global_addr lo
    g0[3] = (unsigned)((ga >> 32) & 0x01ffffffu)       // global_addr [56:32]
          | (2u << 30);                                // type=2 ("image")
    const unsigned td0 = 32u,  td1 = 256u;             // tensor dims (elements)
    const unsigned tile0 = 32u, tile1 = 256u;          // tile dims
    const unsigned st0 = (unsigned)strideK;            // tensor_dim0_stride
    i32x8 g1;
    g1[0] = (int)((1u << 16)                           // data_size=1 (2 bytes)
                | (1u << 20)                           // pad_enable
                | (3u << 22)                           // pad_interval: 16 dwords
                | (3u << 25));                         // pad_amount:   4 dwords
    g1[1] = (int)((td0 & 0xffffu) << 16);              // abar=0 | tensor_dim0 lo
    g1[2] = (int)(((td0 >> 16) & 0xffffu) | ((td1 & 0xffffu) << 16));
    g1[3] = (int)(((td1 >> 16) & 0xffffu) | (tile0 << 16));
    g1[4] = (int)tile1;                                // tile_dim1 | tile_dim2=0
    g1[5] = (int)st0;                                  // dim0 stride [31:0]
    g1[6] = 0;                                         // stride hi | dim1 stride lo
    g1[7] = 0;
    i32x4 gz = {0, 0, 0, 0};
#if __clang_major__ >= 23
    i32x8 gz8 = {0, 0, 0, 0, 0, 0, 0, 0};
    __builtin_amdgcn_tensor_load_to_lds(g0, g1, gz, gz, gz8, 0);
#else
    __builtin_amdgcn_tensor_load_to_lds(g0, g1, gz, gz, 0);
#endif
}

// =====================================================================
// WMMA GEMM: C[M,N] = A[M,K](fp32) @ B(f16, stored transposed [N][K]).
// grid = (N/256, M/16); block = 256 threads = 8 waves, all sharing tileM.
// Per K-step the TDM streams a 256x32 B^T slab into LDS (double buffered,
// padded rows); each wave reads its two 32x16 B fragments as ds_load_b128
// pairs and issues two v_wmma_f32_16x16x32_f16.
// a_mode 0: A row = m
// a_mode 1: A row = nodeidx[m], scaled by nmask[nodeidx[m]]   (feat gather)
// a_mode 2: A row = (m/64)*NPG + turn*64 + (m%64)             (active rows)
// =====================================================================
__global__ __launch_bounds__(256)
void wmma_gemm_f16(const float* __restrict__ A, const _Float16* __restrict__ BhT,
                   float* __restrict__ C, int M, int N, int K,
                   int lda, int ldc,
                   int a_mode, int turn,
                   const int* __restrict__ nodeidx, const float* __restrict__ nmask)
{
    __shared__ _Float16 bstage[2][256 * BSTRIDE_ + 16];   // 2 x ~20 KB (padded rows)

    const int lane  = threadIdx.x & 31;
    const int wave  = threadIdx.x >> 5;
    const int tileM = blockIdx.y;
    const int panel = blockIdx.x << 8;                 // 256-wide column panel
    const int mloc  = lane & 15;
    const int khalf = lane >> 4;

    // A row for this lane (16-bit A layout: both lane halves hold M=0..15)
    const int mg = tileM * 16 + mloc;
    int rowA; float scale = 1.0f;
    if (a_mode == 1)      { int nd = nodeidx[mg]; rowA = nd; scale = nmask[nd]; }
    else if (a_mode == 2) { rowA = (mg >> 6) * NPG_ + turn * 64 + (mg & 63); }
    else                  { rowA = mg; }
    const float* arow = A + (long)rowA * lda;

    const int nloc0 = (wave << 5) + (lane & 15);       // panel-local column, tile 0
    const int nloc1 = nloc0 + 16;                      // panel-local column, tile 1

    const int nk = K >> 5;
    v8f c0 = {}; v8f c1 = {};

    // prologue: prime buffer 0  (B^T row n starts at n*K; slab k-offset 0)
    if (wave == 0)
        tdm_load_slab(BhT + (long)panel * K, (unsigned)(uintptr_t)&bstage[0][0], K);

    for (int ks = 0; ks < nk; ++ks) {
        const int cur = ks & 1;
        const int k0  = ks << 5;
        if (wave == 0) {
            if (ks + 1 < nk) {
                tdm_load_slab(BhT + (long)panel * K + (k0 + 32),
                              (unsigned)(uintptr_t)&bstage[cur ^ 1][0], K);
                __builtin_amdgcn_s_wait_tensorcnt(1);  // current slab landed
            } else {
                __builtin_amdgcn_s_wait_tensorcnt(0);
            }
        }
        __syncthreads();                               // slab visible to all waves

        if (k0 + 32 < K) __builtin_prefetch(arow + k0 + 32, 0, 3);

        // ---- A fragment 16x32 f16: element e -> K = (e/8)*16 + khalf*8 + e%8
        v16h a;
        {
            const float4* p0 = reinterpret_cast<const float4*>(arow + k0 + khalf * 8);
            const float4* p1 = reinterpret_cast<const float4*>(arow + k0 + 16 + khalf * 8);
            float4 f0 = p0[0], f1 = p0[1], f2 = p1[0], f3 = p1[1];
            a[0]=f2h(f0.x*scale);  a[1]=f2h(f0.y*scale);  a[2]=f2h(f0.z*scale);  a[3]=f2h(f0.w*scale);
            a[4]=f2h(f1.x*scale);  a[5]=f2h(f1.y*scale);  a[6]=f2h(f1.z*scale);  a[7]=f2h(f1.w*scale);
            a[8]=f2h(f2.x*scale);  a[9]=f2h(f2.y*scale);  a[10]=f2h(f2.z*scale); a[11]=f2h(f2.w*scale);
            a[12]=f2h(f3.x*scale); a[13]=f2h(f3.y*scale); a[14]=f2h(f3.z*scale); a[15]=f2h(f3.w*scale);
        }
        // ---- B fragments 32x16 f16 from LDS:
        //      element e -> K = (e/8)*16 + khalf*8 + e%8  ==> two ds_load_b128 each
        const _Float16* bb = &bstage[cur][0];
        {
            const int o0 = nloc0 * BSTRIDE_ + (khalf << 3);
            const int o1 = nloc1 * BSTRIDE_ + (khalf << 3);
            v8h l0 = *(const v8h*)(bb + o0);
            v8h h0 = *(const v8h*)(bb + o0 + 16);
            v8h l1 = *(const v8h*)(bb + o1);
            v8h h1 = *(const v8h*)(bb + o1 + 16);
            v16h b0 = __builtin_shufflevector(l0, h0, 0,1,2,3,4,5,6,7,8,9,10,11,12,13,14,15);
            v16h b1 = __builtin_shufflevector(l1, h1, 0,1,2,3,4,5,6,7,8,9,10,11,12,13,14,15);
            c0 = __builtin_amdgcn_wmma_f32_16x16x32_f16(false, a, false, b0, (short)0, c0, false, false);
            c1 = __builtin_amdgcn_wmma_f32_16x16x32_f16(false, a, false, b1, (short)0, c1, false, false);
        }
        __syncthreads();                               // done reading before overwrite
    }

    // ---- C store: VGPR r -> M = tileM*16 + r + 8*khalf, N = panel + nloc
    const int mBase = tileM * 16 + 8 * khalf;
    #pragma unroll
    for (int r = 0; r < 8; ++r) {
        C[(long)(mBase + r) * ldc + panel + nloc0] = c0[r];
        C[(long)(mBase + r) * ldc + panel + nloc1] = c1[r];
    }
}

// =====================================================================
// One-time weight packing (all GEMM B operands -> f16, TRANSPOSED [N][K]):
//   WinT [256][768]  : WinT[n][k]      = f16(Win[k][n])
//   WgRT [256][256]  : WgRT[h*64+f][d] = f16(Wg[h][d][f])
//   WcatT[768][256]  : WcatT[g*256+cc][d] = f16({Wr,Wz,Wn}[g][d][cc])
//   UcatT[768][256]  : same with {Ur,Uz,Un}
// =====================================================================
__global__ __launch_bounds__(256)
void pack_weights(const float* __restrict__ Win, const float* __restrict__ Wg,
                  const float* __restrict__ Wr, const float* __restrict__ Wz, const float* __restrict__ Wn,
                  const float* __restrict__ Ur, const float* __restrict__ Uz, const float* __restrict__ Un,
                  _Float16* __restrict__ WinT, _Float16* __restrict__ WgRT,
                  _Float16* __restrict__ WcatT, _Float16* __restrict__ UcatT)
{
    int i = blockIdx.x * 256 + threadIdx.x;
    if (i < 256 * 256) {
        int c = i >> 8, d = i & 255, h = c >> 6, f = c & 63;
        WgRT[i] = f2h(Wg[h * (NHID_ * 64) + d * 64 + f]);
    }
    if (i < 256 * 768) {
        { int n = i / 768, k = i % 768;
          WinT[i] = f2h(Win[k * NHID_ + n]); }
        { int c = i >> 8, d = i & 255, g = c >> 8, cc = c & 255;
          const float* Ws = (g == 0) ? Wr : (g == 1) ? Wz : Wn;
          const float* Us = (g == 0) ? Ur : (g == 1) ? Uz : Un;
          WcatT[i] = f2h(Ws[d * NHID_ + cc]);
          UcatT[i] = f2h(Us[d * NHID_ + cc]); }
    }
}

// ===================== es/ed = per-head attention logits =============
__global__ __launch_bounds__(256)
void attn_logits_kernel(const float* __restrict__ z,
                        const float* __restrict__ a_src, const float* __restrict__ a_dst,
                        float* __restrict__ es, float* __restrict__ ed)
{
    int i = blockIdx.x * 256 + threadIdx.x;            // over NN*H
    if (i >= NN_ * H_) return;
    int n = i >> 2, h = i & 3;
    const float* zr = z + (long)n * NHID_ + h * 64;
    float s1 = 0.f, s2 = 0.f;
    #pragma unroll 8
    for (int f = 0; f < 64; ++f) {
        float v = zr[f];
        s1 += v * a_src[h * 64 + f];
        s2 += v * a_dst[h * 64 + f];
    }
    es[i] = s1; ed[i] = s2;
}

// ===== per-active-destination softmax + weighted message + ex ========
// Block = one active node (256 threads). Edges of node n are n*32..n*32+31.
__global__ __launch_bounds__(256)
void attn_msg_kernel(const float* __restrict__ z,
                     const float* __restrict__ es, const float* __restrict__ ed,
                     const int* __restrict__ srcArr, int turn,
                     float* __restrict__ ex, float* __restrict__ msg)
{
    const int an   = blockIdx.x;                              // 0..2047
    const int node = (an >> 6) * NPG_ + turn * 64 + (an & 63);
    __shared__ float eraw[DEG_][H_];
    __shared__ float alpha[DEG_][H_];
    __shared__ int   lsrc[DEG_];
    const int t = threadIdx.x;

    if (t < DEG_) lsrc[t] = srcArr[node * DEG_ + t];
    __syncthreads();

    if (t < DEG_ * H_) {
        int e = t >> 2, h = t & 3;
        float v = es[lsrc[e] * H_ + h] + ed[node * H_ + h];
        eraw[e][h] = (v > 0.f) ? v : SLOPE_ * v;              // leaky_relu
    }
    __syncthreads();

    if (t < H_) {                                             // per-head stable softmax
        int h = t;
        float mx = -1e30f;
        for (int e = 0; e < DEG_; ++e) mx = fmaxf(mx, eraw[e][h]);
        float den = 0.f;
        for (int e = 0; e < DEG_; ++e) { float ee = expf(eraw[e][h] - mx); alpha[e][h] = ee; den += ee; }
        float inv = 1.f / (den + 1e-9f);
        for (int e = 0; e < DEG_; ++e) alpha[e][h] *= inv;
    }
    __syncthreads();

    if (t < DEG_)
        ex[node * DEG_ + t] = 0.25f * (alpha[t][0] + alpha[t][1] + alpha[t][2] + alpha[t][3]);

    // msg[an, c] = sum_e alpha[e][c/64] * z[src_e, c]   (coalesced over c)
    const int c = t, h = c >> 6;
    float acc = 0.f;
    #pragma unroll 4
    for (int e = 0; e < DEG_; ++e)
        acc += alpha[e][h] * z[(long)lsrc[e] * NHID_ + c];
    msg[(long)an * NHID_ + c] = acc;
}

// ===================== fused GRU cell update =========================
__global__ __launch_bounds__(256)
void gru_kernel(const float* __restrict__ Xw, const float* __restrict__ Sh,
                const float* __restrict__ br, const float* __restrict__ bz, const float* __restrict__ bn,
                float* __restrict__ s, int turn)
{
    int i = blockIdx.x * 256 + threadIdx.x;            // over ACT_*NHID_
    int r = i >> 8, c = i & 255;
    int node = (r >> 6) * NPG_ + turn * 64 + (r & 63);
    const long o = (long)r * 768;
    float xr = Xw[o + c],       hr = Sh[o + c];
    float xz = Xw[o + 256 + c], hz = Sh[o + 256 + c];
    float xn = Xw[o + 512 + c], hn = Sh[o + 512 + c];
    float rg = 1.f / (1.f + expf(-(xr + hr + br[c])));
    float zg = 1.f / (1.f + expf(-(xz + hz + bz[c])));
    float ng = tanhf(xn + rg * hn + bn[c]);
    float so = s[(long)node * NHID_ + c];
    s[(long)node * NHID_ + c] = (1.f - zg) * ng + zg * so;
}

// ============= attn_score = segment_sum(ex, dst) (contiguous) ========
__global__ __launch_bounds__(256)
void attn_score_kernel(const float* __restrict__ ex, float* __restrict__ attn)
{
    int n = blockIdx.x * 256 + threadIdx.x;
    if (n >= NN_) return;
    float a = 0.f;
    #pragma unroll
    for (int e = 0; e < DEG_; ++e) a += ex[(long)n * DEG_ + e];
    attn[n] = a;
}

// ====== per (graph, uid) top-3 by attn among parity-matched nodes ====
__global__ __launch_bounds__(256)
void topk_kernel(const float* __restrict__ s, const float* __restrict__ attn,
                 const int* __restrict__ tidArr, float* __restrict__ out)
{
    const int b = blockIdx.x >> 1, u = blockIdx.x & 1;
    __shared__ int idx3[3];
    if (threadIdx.x == 0) {
        int c0 = -1, c1 = -1;
        for (int kk = 0; kk < 3; ++kk) {
            float best = -3.0e38f; int bi = 0;
            for (int p = 0; p < NPG_; ++p) {
                if ((tidArr[b * NPG_ + p] & 1) != u) continue;   // uid = tid % 2
                if (p == c0 || p == c1) continue;
                float sc = attn[b * NPG_ + p];
                if (sc > best) { best = sc; bi = p; }            // ties -> lowest index
            }
            idx3[kk] = bi;
            if (kk == 0) c0 = bi; else if (kk == 1) c1 = bi;
        }
    }
    __syncthreads();
    float* base = out + (long)u * (B_ * 3 * NHID_) + (long)b * (3 * NHID_);
    for (int j = threadIdx.x; j < 3 * NHID_; j += 256) {
        int kk = j >> 8, c = j & 255;
        base[j] = s[((long)b * NPG_ + idx3[kk]) * NHID_ + c];
    }
}

// =====================================================================
extern "C" void kernel_launch(void* const* d_in, const int* in_sizes, int n_in,
                              void* d_out, int out_size, void* d_ws, size_t ws_size,
                              hipStream_t stream)
{
    (void)in_sizes; (void)n_in; (void)out_size; (void)ws_size;
    const float* emb     = (const float*)d_in[0];
    const float* nmask   = (const float*)d_in[1];
    const int*   nodeidx = (const int*)  d_in[2];
    const int*   src     = (const int*)  d_in[3];
    /* d_in[4] = dst is implicit: dst[e] = e / 32 */
    const int*   tid     = (const int*)  d_in[5];
    const float* Win     = (const float*)d_in[6];
    const float* Wg      = (const float*)d_in[7];
    const float* a_src   = (const float*)d_in[8];
    const float* a_dst   = (const float*)d_in[9];
    const float* Wr = (const float*)d_in[10];
    const float* Ur = (const float*)d_in[11];
    const float* Wz = (const float*)d_in[12];
    const float* Uz = (const float*)d_in[13];
    const float* Wn = (const float*)d_in[14];
    const float* Un = (const float*)d_in[15];
    const float* br = (const float*)d_in[16];
    const float* bz = (const float*)d_in[17];
    const float* bn = (const float*)d_in[18];
    float* out = (float*)d_out;

    // -------- workspace carve-up (~42 MB fp32 + ~1.3 MB f16) --------
    float* w    = (float*)d_ws;
    float* s    = w; w += (long)NN_ * NHID_;
    float* z    = w; w += (long)NN_ * NHID_;
    float* es   = w; w += (long)NN_ * H_;
    float* ed   = w; w += (long)NN_ * H_;
    float* ex   = w; w += (long)NN_ * DEG_;
    float* msg  = w; w += (long)ACT_ * NHID_;
    float* Xw   = w; w += (long)ACT_ * 768;
    float* Sh   = w; w += (long)ACT_ * 768;
    float* attn = w; w += NN_;
    _Float16* hb    = (_Float16*)w;
    _Float16* WinT  = hb; hb += 256 * 768;
    _Float16* WgRT  = hb; hb += 256 * 256;
    _Float16* WcatT = hb; hb += 768 * 256;
    _Float16* UcatT = hb; hb += 768 * 256;

    pack_weights<<<768, 256, 0, stream>>>(Win, Wg, Wr, Wz, Wn, Ur, Uz, Un,
                                          WinT, WgRT, WcatT, UcatT);

    // s = (emb * nmask)[nodeidx] @ Win   [12288,768]x[768,256]
    wmma_gemm_f16<<<dim3(NHID_/256, NN_/16), 256, 0, stream>>>(
        emb, WinT, s, NN_, NHID_, NFEAT_, NFEAT_, NHID_, 1, 0, nodeidx, nmask);

    for (int t = 0; t < T_; ++t) {
        // z = s @ WgR   [12288,256]x[256,256]
        wmma_gemm_f16<<<dim3(NHID_/256, NN_/16), 256, 0, stream>>>(
            s, WgRT, z, NN_, NHID_, NHID_, NHID_, NHID_, 0, 0, nullptr, nullptr);
        attn_logits_kernel<<<(NN_*H_)/256, 256, 0, stream>>>(z, a_src, a_dst, es, ed);
        attn_msg_kernel<<<ACT_, 256, 0, stream>>>(z, es, ed, src, t, ex, msg);
        // Xw = msg @ [Wr|Wz|Wn], Sh = s_active @ [Ur|Uz|Un]   [2048,256]x[256,768]
        wmma_gemm_f16<<<dim3(768/256, ACT_/16), 256, 0, stream>>>(
            msg, WcatT, Xw, ACT_, 768, NHID_, NHID_, 768, 0, 0, nullptr, nullptr);
        wmma_gemm_f16<<<dim3(768/256, ACT_/16), 256, 0, stream>>>(
            s, UcatT, Sh, ACT_, 768, NHID_, NHID_, 768, 2, t, nullptr, nullptr);
        gru_kernel<<<(ACT_*NHID_)/256, 256, 0, stream>>>(Xw, Sh, br, bz, bn, s, t);
    }

    attn_score_kernel<<<NN_/256, 256, 0, stream>>>(ex, attn);
    topk_kernel<<<2*B_, 256, 0, stream>>>(s, attn, tid, out);
}
